// NetFighterCritic_76897094468415
// MI455X (gfx1250) — compile-verified, matrix operand order checked by reading, not run on hardware
//
#include <hip/hip_runtime.h>
#include <hip/hip_bf16.h>

typedef __attribute__((ext_vector_type(16))) _Float16 v16h;
typedef __attribute__((ext_vector_type(8)))  float    v8f;

#define NAGENT 64
#define NHEAD  4
#define CONV_COLS 120   // staged input width: px(0..111)+kx(0..4)+2 pad

// ---------------------------------------------------------------------------
// CDNA5 async global->LDS copy (ASYNCcnt path), guarded so either toolchain
// compiles.  Falls back to plain VGPR staging when the builtin is absent.
// The builtin's parameters are v4i* in AS1 (global) / AS3 (LDS).
// ---------------------------------------------------------------------------
#if __has_builtin(__builtin_amdgcn_global_load_async_to_lds_b128)
#define HAVE_ASYNC_LDS 1
typedef int v4i __attribute__((vector_size(16)));
typedef __attribute__((address_space(1))) v4i* gptr_v4i;
typedef __attribute__((address_space(3))) v4i* lptr_v4i;
__device__ inline void async_copy_f4(const float* g, float* l) {
    __builtin_amdgcn_global_load_async_to_lds_b128(
        (gptr_v4i)g, (lptr_v4i)l, 0, 0);
}
__device__ inline void async_wait0() {
#if __has_builtin(__builtin_amdgcn_s_wait_asynccnt)
    __builtin_amdgcn_s_wait_asynccnt(0);
#else
    asm volatile("s_wait_asynccnt 0" ::: "memory");
#endif
}
#else
#define HAVE_ASYNC_LDS 0
#endif

// Swizzled LDS index for 16-bit WMMA fragments, K-tile of 32:
//   frag pos p (0..15) of lane l maps to k = (p>>3)*16 + (l>=16?8:0) + (p&7)
//   => store element (row, k) at [row][half=(k>>3)&1][pos=((k>>4)<<3)|(k&7)]
__device__ inline int swz(int row, int k) {
    return ((row * 2 + ((k >> 3) & 1)) << 4) + (((k >> 4) << 3) | (k & 7));
}

__device__ inline float block_reduce_sum(float v, float* red, int tid, int nthreads) {
    red[tid] = v;
    __syncthreads();
    for (int s = nthreads >> 1; s > 0; s >>= 1) {
        if (tid < s) red[tid] += red[tid + s];
        __syncthreads();
    }
    float r = red[0];
    __syncthreads();
    return r;
}

// ---------------------------------------------------------------------------
// Kernel 1: per-agent 5x5x5 conv (pad 2) as implicit GEMM on WMMA f16
// (K=125 padded to 128 -> 4 chained 16x16x32 WMMAs per 16-pixel tile),
// fused bias+ReLU+4x4 maxpool.  Block = (agent, path, pool_row), 4 waves,
// each wave computes one conv row as 7 tiles of 16 pixels.
// ---------------------------------------------------------------------------
__global__ __launch_bounds__(128)
void conv_pool_kernel(const float* __restrict__ imgs,
                      const float* __restrict__ conv1_w, const float* __restrict__ conv1_b,
                      const float* __restrict__ conv2_w, const float* __restrict__ conv2_b,
                      float* __restrict__ pooled /* [A][2][5000] */)
{
    const int agent = blockIdx.x, path = blockIdx.y, prow = blockIdx.z;
    const int tid = threadIdx.x;
    const int wave = tid >> 5, lane = tid & 31;

    __shared__ __attribute__((aligned(32))) _Float16 wlds[4 * 16 * 32];  // swizzled [ks][m][k]
    __shared__ _Float16 inlds[5 * 8 * CONV_COLS];                        // [c][row][col]
    __shared__ float    convlds[8 * 4 * 100];                            // [ch][wave_row][px]

    const float* cw = (path == 0 ? conv1_w : conv2_w) + agent * 8 * 125;
    const float* cb = (path == 0 ? conv1_b : conv2_b) + agent * 8;

    // stage weights pre-swizzled (pad M 8->16, K 125->128 with zeros)
    for (int i = tid; i < 16 * 128; i += 128) {
        int mm = i >> 7, k = i & 127;
        float v = (mm < 8 && k < 125) ? cw[mm * 125 + k] : 0.f;
        int ks = k >> 5, kk = k & 31;
        wlds[swz(ks * 16 + mm, kk)] = (_Float16)v;
    }
    // stage input rows py0-2 .. py0+5, cols -2..117 (zero padded)
    const int py0 = prow * 4;
    for (int i = tid; i < 5 * 8 * CONV_COLS; i += 128) {
        int c   = i / (8 * CONV_COLS);
        int rem = i % (8 * CONV_COLS);
        int r = rem / CONV_COLS, col = rem % CONV_COLS;
        int iy = py0 + r - 2, ix = col - 2;
        float v = 0.f;
        if (iy >= 0 && iy < 100 && ix >= 0 && ix < 100)
            v = imgs[((agent * 5 + c) * 100 + iy) * 100 + ix];
        inlds[i] = (_Float16)v;
    }
    __syncthreads();

    const int m  = lane & 15;   // A row / B column within the 16-wide tile
    const int hi = lane >> 4;   // K-half select per CDNA5 16-bit operand layout
    v16h afrag[4];
#pragma unroll
    for (int ks = 0; ks < 4; ++ks)
        afrag[ks] = *(const v16h*)&wlds[(ks * 16 + m) * 32 + hi * 16];

    float bias[8];
#pragma unroll
    for (int r = 0; r < 8; ++r) bias[r] = cb[r];

    const int w = wave;   // conv row within the pool window
#pragma unroll 1
    for (int tile = 0; tile < 7; ++tile) {
        const int px = tile * 16 + m;   // this lane's output pixel column
        v8f acc = {};
#pragma unroll
        for (int ks = 0; ks < 4; ++ks) {
            v16h bfrag;
#pragma unroll
            for (int p = 0; p < 16; ++p) {
                int k = ks * 32 + ((p >> 3) << 4) + hi * 8 + (p & 7);
                _Float16 v = (_Float16)0.f;
                if (k < 125) {
                    int c = k / 25, rr = k % 25;
                    int ky = rr / 5, kx = rr % 5;
                    v = inlds[(c * 8 + (w + ky)) * CONV_COLS + (px + kx)];
                }
                bfrag[p] = v;
            }
            acc = __builtin_amdgcn_wmma_f32_16x16x32_f16(false, afrag[ks], false, bfrag,
                                                         (short)0, acc, false, false);
        }
        // D layout: lanes 0..15 hold M=0..7 (the 8 real channels); N = lane
        if (lane < 16 && px < 100) {
#pragma unroll
            for (int r = 0; r < 8; ++r) {
                float v = acc[r] + bias[r];
                v = v > 0.f ? v : 0.f;
                convlds[(r * 4 + w) * 100 + px] = v;
            }
        }
    }
    __syncthreads();

    // 4x4 maxpool over the 4 conv rows
    for (int i = tid; i < 8 * 25; i += 128) {
        int ch = i / 25, ox = i % 25;
        float mx = -INFINITY;
#pragma unroll
        for (int r = 0; r < 4; ++r)
#pragma unroll
            for (int cidx = 0; cidx < 4; ++cidx) {
                float v = convlds[(ch * 4 + r) * 100 + ox * 4 + cidx];
                mx = v > mx ? v : mx;
            }
        pooled[(agent * 2 + path) * 5000 + ch * 625 + prow * 25 + ox] = mx;
    }
}

// ---------------------------------------------------------------------------
// Kernel 2: per-agent LN(5000) -> GEMV 5000->256 (bandwidth streaming) ->
// LN(256) -> tanh.  One block per agent; 256 threads = 4 d-chunks x 64 quads.
// Activation staging uses async global->LDS when available.
// ---------------------------------------------------------------------------
__global__ __launch_bounds__(256)
void img_fc_kernel(const float* __restrict__ pooled,
                   const float* __restrict__ ln1g, const float* __restrict__ ln1b,
                   const float* __restrict__ fcw,  const float* __restrict__ fcb,
                   const float* __restrict__ ln2g, const float* __restrict__ ln2b,
                   float* __restrict__ dst /* [A][512], writes cols 0..255 */,
                   int path)
{
    const int agent = blockIdx.x, tid = threadIdx.x;
    __shared__ __attribute__((aligned(16))) float xln[5000];
    __shared__ float part[4 * 256];
    __shared__ float red[256];

    const float* x = pooled + (agent * 2 + path) * 5000;
    float s = 0.f, s2 = 0.f;
#if HAVE_ASYNC_LDS
    for (int base = tid * 4; base < 5000; base += 1024)
        async_copy_f4(x + base, &xln[base]);
    async_wait0();
    __syncthreads();
    for (int d = tid; d < 5000; d += 256) {
        float v = xln[d];
        s += v; s2 += v * v;
    }
#else
    for (int d = tid; d < 5000; d += 256) {
        float v = x[d];
        s += v; s2 += v * v; xln[d] = v;
    }
#endif
    s  = block_reduce_sum(s,  red, tid, 256);
    s2 = block_reduce_sum(s2, red, tid, 256);
    const float mean = s / 5000.f;
    const float rstd = rsqrtf(s2 / 5000.f - mean * mean + 1e-5f);
    for (int d = tid; d < 5000; d += 256)
        xln[d] = (xln[d] - mean) * rstd * ln1g[agent * 5000 + d] + ln1b[agent * 5000 + d];
    __syncthreads();

    // GEMV: thread (dc, j) -> outputs o=4j..4j+3, d-range chunk dc
    const int j = tid & 63, dc = tid >> 6, o = j * 4;
    const float* wb = fcw + (size_t)agent * 5000 * 256;
    float4 acc = make_float4(0.f, 0.f, 0.f, 0.f);
    for (int d = dc * 1250; d < (dc + 1) * 1250; ++d) {
        const float xv = xln[d];
        const float4 w4 = *(const float4*)(wb + (size_t)d * 256 + o);
        acc.x += xv * w4.x; acc.y += xv * w4.y;
        acc.z += xv * w4.z; acc.w += xv * w4.w;
    }
    part[dc * 256 + o + 0] = acc.x; part[dc * 256 + o + 1] = acc.y;
    part[dc * 256 + o + 2] = acc.z; part[dc * 256 + o + 3] = acc.w;
    __syncthreads();

    float y[4] = {0.f, 0.f, 0.f, 0.f};
    float ys = 0.f, ys2 = 0.f;
    if (dc == 0) {
#pragma unroll
        for (int c = 0; c < 4; ++c) {
            float v = fcb[agent * 256 + o + c];
#pragma unroll
            for (int q = 0; q < 4; ++q) v += part[q * 256 + o + c];
            y[c] = v; ys += v; ys2 += v * v;
        }
    }
    ys  = block_reduce_sum(ys,  red, tid, 256);
    ys2 = block_reduce_sum(ys2, red, tid, 256);
    const float m2 = ys / 256.f;
    const float r2 = rsqrtf(ys2 / 256.f - m2 * m2 + 1e-5f);
    if (dc == 0) {
#pragma unroll
        for (int c = 0; c < 4; ++c) {
            float v = (y[c] - m2) * r2 * ln2g[agent * 256 + o + c] + ln2b[agent * 256 + o + c];
            dst[agent * 512 + o + c] = tanhf(v);
        }
    }
}

// ---------------------------------------------------------------------------
// Kernel 3: info MLPs (small, per agent).  Fills all_enc[:,256:512] and
// s_enc[:,256:512].
// ---------------------------------------------------------------------------
__global__ __launch_bounds__(256)
void info_kernel(const float* __restrict__ infos, const float* __restrict__ acts,
                 const float* __restrict__ fc1w, const float* __restrict__ fc1b,
                 const float* __restrict__ ln1g, const float* __restrict__ ln1b,
                 const float* __restrict__ fc2w, const float* __restrict__ fc2b,
                 const float* __restrict__ ln2g, const float* __restrict__ ln2b,
                 const float* __restrict__ i2w,  const float* __restrict__ i2b,
                 const float* __restrict__ i2lg, const float* __restrict__ i2lb,
                 float* __restrict__ all_enc, float* __restrict__ s_enc)
{
    const int agent = blockIdx.x, tid = threadIdx.x;
    __shared__ float comb[259];
    __shared__ float h[128];
    __shared__ float red[256];

    if (tid < 259)
        comb[tid] = (tid < 3) ? infos[agent * 3 + tid] : acts[agent * 256 + (tid - 3)];
    __syncthreads();

    // fc1: 259 -> 128, LN(128), tanh
    float hv = 0.f;
    if (tid < 128) {
        for (int d = 0; d < 259; ++d)
            hv += comb[d] * fc1w[(agent * 259 + d) * 128 + tid];
        hv += fc1b[agent * 128 + tid];
    }
    float c1 = (tid < 128) ? hv : 0.f;
    float c2 = (tid < 128) ? hv * hv : 0.f;
    float s  = block_reduce_sum(c1, red, tid, 256);
    float s2 = block_reduce_sum(c2, red, tid, 256);
    float mean = s / 128.f, rstd = rsqrtf(s2 / 128.f - mean * mean + 1e-5f);
    if (tid < 128)
        h[tid] = tanhf((hv - mean) * rstd * ln1g[agent * 128 + tid] + ln1b[agent * 128 + tid]);
    __syncthreads();

    // fc2: 128 -> 256, LN(256), relu -> info1 -> all_enc[:,256+o]
    float iv = 0.f;
    for (int k = 0; k < 128; ++k)
        iv += h[k] * fc2w[(agent * 128 + k) * 256 + tid];
    iv += fc2b[agent * 256 + tid];
    s  = block_reduce_sum(iv, red, tid, 256);
    s2 = block_reduce_sum(iv * iv, red, tid, 256);
    mean = s / 256.f; rstd = rsqrtf(s2 / 256.f - mean * mean + 1e-5f);
    float info1 = (iv - mean) * rstd * ln2g[agent * 256 + tid] + ln2b[agent * 256 + tid];
    all_enc[agent * 512 + 256 + tid] = info1 > 0.f ? info1 : 0.f;

    // info2: 3 -> 256, LN(256), tanh -> s_enc[:,256+o]
    float ov = i2b[agent * 256 + tid];
#pragma unroll
    for (int d = 0; d < 3; ++d)
        ov += infos[agent * 3 + d] * i2w[(agent * 3 + d) * 256 + tid];
    s  = block_reduce_sum(ov, red, tid, 256);
    s2 = block_reduce_sum(ov * ov, red, tid, 256);
    mean = s / 256.f; rstd = rsqrtf(s2 / 256.f - mean * mean + 1e-5f);
    s_enc[agent * 512 + 256 + tid] =
        tanhf((ov - mean) * rstd * i2lg[agent * 256 + tid] + i2lb[agent * 256 + tid]);
}

// ---------------------------------------------------------------------------
// Kernel 4: generic batched WMMA-f16 GEMM:  C[b] = A(64xK) * B(KxN) (+bias)
// with pre-swizzled f32->f16 LDS staging so each fragment is one contiguous
// 32B LDS read (2x ds_load_b128).  128 threads = 4 waves; 64x64 output tile.
// ---------------------------------------------------------------------------
__global__ __launch_bounds__(128)
void gemm64_wmma(const float* __restrict__ Abase, long long strideA,
                 const float* __restrict__ Bbase, long long strideB,
                 const float* __restrict__ biasBase, long long strideBias,
                 float* __restrict__ Cbase, long long strideC,
                 int N, int K, int transB, int leaky, float scale)
{
    const int bh = blockIdx.y;
    const float* Ap = Abase + (size_t)bh * strideA;
    const float* Bp = Bbase + (size_t)bh * strideB;
    float*       Cp = Cbase + (size_t)bh * strideC;
    const int n0 = blockIdx.x * 64;
    const int tid = threadIdx.x, wave = tid >> 5, lane = tid & 31;

    __shared__ __attribute__((aligned(32))) _Float16 aLds[64 * 32];  // swizzled [row][k]
    __shared__ __attribute__((aligned(32))) _Float16 bLds[64 * 32];  // swizzled [n][k]
    v8f acc[4] = {};

    const int nl = lane & 15;
    const int hi = lane >> 4;

    for (int k0 = 0; k0 < K; k0 += 32) {
        __syncthreads();
        for (int i = tid; i < 64 * 32; i += 128) {
            int r = i >> 5, c = i & 31, kk = k0 + c;
            float v = (kk < K) ? Ap[(size_t)r * K + kk] : 0.f;
            aLds[swz(r, c)] = (_Float16)v;
        }
        for (int i = tid; i < 32 * 64; i += 128) {
            int c = i >> 6, nn = i & 63;
            int kk = k0 + c, gn = n0 + nn;
            float v = 0.f;
            if (kk < K && gn < N)
                v = transB ? Bp[(size_t)gn * K + kk] : Bp[(size_t)kk * N + gn];
            bLds[swz(nn, c)] = (_Float16)v;
        }
        __syncthreads();

        const v16h bfrag = *(const v16h*)&bLds[((wave * 16 + nl) * 2 + hi) * 16];
#pragma unroll
        for (int mt = 0; mt < 4; ++mt) {
            const v16h afrag = *(const v16h*)&aLds[((mt * 16 + nl) * 2 + hi) * 16];
            acc[mt] = __builtin_amdgcn_wmma_f32_16x16x32_f16(false, afrag, false, bfrag,
                                                             (short)0, acc[mt], false, false);
        }
    }

    const float* biasP = biasBase ? biasBase + (size_t)bh * strideBias : nullptr;
    const int n = n0 + wave * 16 + nl;
    const int mb = hi * 8;
    if (n < N) {
#pragma unroll
        for (int mt = 0; mt < 4; ++mt)
#pragma unroll
            for (int r = 0; r < 8; ++r) {
                float v = acc[mt][r] * scale;
                if (biasP) v += biasP[n];
                if (leaky) v = v > 0.f ? v : 0.01f * v;
                Cp[(size_t)(mt * 16 + mb + r) * N + n] = v;
            }
    }
}

// ---------------------------------------------------------------------------
// Kernel 5: masked softmax over rows of logits [H,64,64], diagonal = -inf.
// ---------------------------------------------------------------------------
__global__ __launch_bounds__(64)
void softmax_mask_kernel(const float* __restrict__ logits, float* __restrict__ probs)
{
    const int row = blockIdx.x;            // h*64 + i
    const int i = row & 63, tid = threadIdx.x;
    __shared__ float red[64];
    float v = logits[row * 64 + tid];
    if (tid == i) v = -INFINITY;
    red[tid] = v; __syncthreads();
    for (int s = 32; s > 0; s >>= 1) {
        if (tid < s) red[tid] = fmaxf(red[tid], red[tid + s]);
        __syncthreads();
    }
    const float mx = red[0]; __syncthreads();
    const float e = __expf(v - mx);
    red[tid] = e; __syncthreads();
    for (int s = 32; s > 0; s >>= 1) {
        if (tid < s) red[tid] += red[tid + s];
        __syncthreads();
    }
    probs[row * 64 + tid] = e / red[0];
}

// ---------------------------------------------------------------------------
// Kernel 6: assemble critic_in = [s_enc | other0 | other1 | other2 | other3]
// ---------------------------------------------------------------------------
__global__ __launch_bounds__(256)
void concat_kernel(const float* __restrict__ s_enc, const float* __restrict__ other,
                   float* __restrict__ critic_in)
{
    const int idx = blockIdx.x * 256 + threadIdx.x;
    if (idx >= NAGENT * 4608) return;
    const int a = idx / 4608, c = idx % 4608;
    float v;
    if (c < 512) v = s_enc[a * 512 + c];
    else {
        const int h = (c - 512) >> 10, e = (c - 512) & 1023;
        v = other[((size_t)h * 64 + a) * 1024 + e];
    }
    critic_in[idx] = v;
}

// ---------------------------------------------------------------------------
// Kernel 7: per-agent critic head: GEMV 4608->256 (+bias, leaky) then
// dot(256) -> scalar.  float4 streaming + prefetch + async LDS staging.
// ---------------------------------------------------------------------------
__global__ __launch_bounds__(256)
void critic_kernel(const float* __restrict__ critic_in,
                   const float* __restrict__ w1, const float* __restrict__ b1,
                   const float* __restrict__ w3, const float* __restrict__ b3,
                   float* __restrict__ out)
{
    const int agent = blockIdx.x, tid = threadIdx.x;
    __shared__ __attribute__((aligned(16))) float xin[4608];
    __shared__ float part[4 * 256];
    __shared__ float red[256];
    __shared__ float hc[256];

#if HAVE_ASYNC_LDS
    for (int base = tid * 4; base < 4608; base += 1024)
        async_copy_f4(critic_in + agent * 4608 + base, &xin[base]);
    async_wait0();
#else
    for (int d = tid; d < 4608; d += 256) xin[d] = critic_in[agent * 4608 + d];
#endif
    __syncthreads();

    const int j = tid & 63, dc = tid >> 6, o = j * 4;
    const float* wb = w1 + (size_t)agent * 4608 * 256;
    float4 acc = make_float4(0.f, 0.f, 0.f, 0.f);
    for (int d = dc * 1152; d < (dc + 1) * 1152; ++d) {
        if ((d & 15) == 0)
            __builtin_prefetch(wb + (size_t)(d + 16) * 256 + o, 0, 1);
        const float xv = xin[d];
        const float4 w4 = *(const float4*)(wb + (size_t)d * 256 + o);
        acc.x += xv * w4.x; acc.y += xv * w4.y;
        acc.z += xv * w4.z; acc.w += xv * w4.w;
    }
    part[dc * 256 + o + 0] = acc.x; part[dc * 256 + o + 1] = acc.y;
    part[dc * 256 + o + 2] = acc.z; part[dc * 256 + o + 3] = acc.w;
    __syncthreads();
    if (dc == 0) {
#pragma unroll
        for (int c = 0; c < 4; ++c) {
            float v = b1[agent * 256 + o + c];
#pragma unroll
            for (int q = 0; q < 4; ++q) v += part[q * 256 + o + c];
            hc[o + c] = v > 0.f ? v : 0.01f * v;
        }
    }
    __syncthreads();
    float p = hc[tid] * w3[agent * 256 + tid];
    p = block_reduce_sum(p, red, tid, 256);
    if (tid == 0) out[agent] = p + b3[agent];
}

// ---------------------------------------------------------------------------
extern "C" void kernel_launch(void* const* d_in, const int* in_sizes, int n_in,
                              void* d_out, int out_size, void* d_ws, size_t ws_size,
                              hipStream_t stream) {
    (void)in_sizes; (void)n_in; (void)out_size; (void)ws_size;
    const float* imgs    = (const float*)d_in[0];
    const float* infos   = (const float*)d_in[1];
    const float* acts    = (const float*)d_in[2];
    const float* conv1_w = (const float*)d_in[3];
    const float* conv1_b = (const float*)d_in[4];
    const float* i1l1g   = (const float*)d_in[5];
    const float* i1l1b   = (const float*)d_in[6];
    const float* i1fw    = (const float*)d_in[7];
    const float* i1fb    = (const float*)d_in[8];
    const float* i1l2g   = (const float*)d_in[9];
    const float* i1l2b   = (const float*)d_in[10];
    const float* conv2_w = (const float*)d_in[11];
    const float* conv2_b = (const float*)d_in[12];
    const float* i2l1g   = (const float*)d_in[13];
    const float* i2l1b   = (const float*)d_in[14];
    const float* i2fw    = (const float*)d_in[15];
    const float* i2fb    = (const float*)d_in[16];
    const float* i2l2g   = (const float*)d_in[17];
    const float* i2l2b   = (const float*)d_in[18];
    const float* nfc1w   = (const float*)d_in[19];
    const float* nfc1b   = (const float*)d_in[20];
    const float* nl1g    = (const float*)d_in[21];
    const float* nl1b    = (const float*)d_in[22];
    const float* nfc2w   = (const float*)d_in[23];
    const float* nfc2b   = (const float*)d_in[24];
    const float* nl2g    = (const float*)d_in[25];
    const float* nl2b    = (const float*)d_in[26];
    const float* n2w     = (const float*)d_in[27];
    const float* n2b     = (const float*)d_in[28];
    const float* n2lg    = (const float*)d_in[29];
    const float* n2lb    = (const float*)d_in[30];
    const float* key_w   = (const float*)d_in[31];
    const float* sel_w   = (const float*)d_in[32];
    const float* val_w   = (const float*)d_in[33];
    const float* val_b   = (const float*)d_in[34];
    const float* c1w     = (const float*)d_in[35];
    const float* c1b     = (const float*)d_in[36];
    const float* c3w     = (const float*)d_in[37];
    const float* c3b     = (const float*)d_in[38];
    float* out = (float*)d_out;

    float* ws        = (float*)d_ws;
    float* pooled    = ws;                     // 64*2*5000   = 640000
    float* all_enc   = pooled    + 640000;     // 64*512      =  32768
    float* s_enc     = all_enc   + 32768;      //             =  32768
    float* keys      = s_enc     + 32768;      // 4*64*1024   = 262144
    float* sels      = keys      + 262144;
    float* vals      = sels      + 262144;
    float* logits    = vals      + 262144;     // 4*64*64     =  16384
    float* probs     = logits    + 16384;
    float* other     = probs     + 16384;      // 4*64*1024   = 262144
    float* critic_in = other     + 262144;     // 64*4608     = 294912

    // 1) conv + pool (both paths), WMMA implicit GEMM
    conv_pool_kernel<<<dim3(NAGENT, 2, 25), 128, 0, stream>>>(
        imgs, conv1_w, conv1_b, conv2_w, conv2_b, pooled);

    // 2) image FC paths -> all_enc[:, :256], s_enc[:, :256]
    img_fc_kernel<<<NAGENT, 256, 0, stream>>>(pooled, i1l1g, i1l1b, i1fw, i1fb,
                                              i1l2g, i1l2b, all_enc, 0);
    img_fc_kernel<<<NAGENT, 256, 0, stream>>>(pooled, i2l1g, i2l1b, i2fw, i2fb,
                                              i2l2g, i2l2b, s_enc, 1);

    // 3) info paths -> all_enc[:, 256:], s_enc[:, 256:]
    info_kernel<<<NAGENT, 256, 0, stream>>>(infos, acts, nfc1w, nfc1b, nl1g, nl1b,
                                            nfc2w, nfc2b, nl2g, nl2b,
                                            n2w, n2b, n2lg, n2lb, all_enc, s_enc);

    // 4) attention projections (WMMA GEMM, batched over heads)
    const long long sW = 512LL * 1024, sC = 64LL * 1024;
    gemm64_wmma<<<dim3(16, NHEAD), 128, 0, stream>>>(
        all_enc, 0LL, key_w, sW, nullptr, 0LL, keys, sC, 1024, 512, 0, 0, 1.0f);
    gemm64_wmma<<<dim3(16, NHEAD), 128, 0, stream>>>(
        s_enc, 0LL, sel_w, sW, nullptr, 0LL, sels, sC, 1024, 512, 0, 0, 1.0f);
    gemm64_wmma<<<dim3(16, NHEAD), 128, 0, stream>>>(
        all_enc, 0LL, val_w, sW, val_b, 1024LL, vals, sC, 1024, 512, 0, 1, 1.0f);

    // 5) logits = sels . keys^T / sqrt(1024)
    gemm64_wmma<<<dim3(1, NHEAD), 128, 0, stream>>>(
        sels, sC, keys, sC, nullptr, 0LL, logits, 64LL * 64, 64, 1024, 1, 0, 0.03125f);

    // 6) masked softmax
    softmax_mask_kernel<<<NHEAD * NAGENT, 64, 0, stream>>>(logits, probs);

    // 7) other = probs . vals  (WMMA, K=64)
    gemm64_wmma<<<dim3(16, NHEAD), 128, 0, stream>>>(
        probs, 64LL * 64, vals, sC, nullptr, 0LL, other, sC, 1024, 64, 0, 0, 1.0f);

    // 8) concat + critic head
    concat_kernel<<<(NAGENT * 4608 + 255) / 256, 256, 0, stream>>>(s_enc, other, critic_in);
    critic_kernel<<<NAGENT, 256, 0, stream>>>(critic_in, c1w, c1b, c3w, c3b, out);
}
